// VQVAE_25752623907357
// MI455X (gfx1250) — matrix operand order repaired and checked by reading, not compiled, not run on hardware
//
#include <hip/hip_runtime.h>
#include <hip/hip_bf16.h>

typedef __attribute__((ext_vector_type(16))) _Float16 v16h;
typedef __attribute__((ext_vector_type(8)))  float    v8f;
typedef __attribute__((ext_vector_type(4)))  int      v4i;

#define TILE_M 128
#define TILE_N 128
#define TILE_K 64
#define LDSROW 72   // padded LDS k-stride in halves (144B = 9*16B; B128-aligned)

#if __has_builtin(__builtin_amdgcn_global_load_async_to_lds_b128)
#define USE_ASYNC_LDS 1
#else
#define USE_ASYNC_LDS 0
#endif

__device__ __forceinline__ void vqvae_async_fence() {
#if USE_ASYNC_LDS
#if __has_builtin(__builtin_amdgcn_s_wait_asynccnt)
    __builtin_amdgcn_s_wait_asynccnt(0);
#else
    asm volatile("s_wait_asynccnt 0" ::: "memory");
#endif
#endif
}

// out = act(A[M,K](f16) @ Bt[N,K](f16)^T + bias); dual outputs: f32 and/or f16.
__global__ __launch_bounds__(256)
void vqvae_gemm_wmma(const _Float16* __restrict__ A,
                     const _Float16* __restrict__ Bt,   // [N,K] (pre-transposed weight)
                     const float* __restrict__ bias,
                     float* __restrict__ out32,
                     _Float16* __restrict__ out16,
                     int M, int N, int K, int relu)
{
    __shared__ _Float16 Ash[2][TILE_M * LDSROW];
    __shared__ _Float16 Bsh[2][TILE_N * LDSROW];

    const int tid  = threadIdx.x;
    const int lane = tid & 31;
    const int wid  = tid >> 5;     // 0..7
    const int mw   = wid >> 2;     // 0..1  -> 64-row band
    const int nw   = wid & 3;      // 0..3  -> 32-col band
    const int lm   = lane & 15;
    const int kh   = lane >> 4;

    const int m0 = blockIdx.y * TILE_M;
    const int n0 = blockIdx.x * TILE_N;

    // Stage one 128x64 f16 tile of A and of Bt into LDS buffer `bufi`.
    auto stage = [&](int k0, int bufi) {
        #pragma unroll
        for (int t = 0; t < 4; ++t) {
            const int chunk = tid + t * 256;       // 1024 chunks of 16B per tile
            const int row   = chunk >> 3;          // 0..127
            const int c     = chunk & 7;           // which 16B of the 128B row
            const _Float16* ga = A  + (size_t)(m0 + row) * K + k0 + c * 8;
            const _Float16* gb = Bt + (size_t)(n0 + row) * K + k0 + c * 8;
            _Float16* la = &Ash[bufi][row * LDSROW + c * 8];
            _Float16* lb = &Bsh[bufi][row * LDSROW + c * 8];
#if USE_ASYNC_LDS
            __builtin_amdgcn_global_load_async_to_lds_b128(
                (__attribute__((address_space(1))) v4i*)ga,
                (__attribute__((address_space(3))) v4i*)la, 0, 0);
            __builtin_amdgcn_global_load_async_to_lds_b128(
                (__attribute__((address_space(1))) v4i*)gb,
                (__attribute__((address_space(3))) v4i*)lb, 0, 0);
#else
            *(uint4*)la = *(const uint4*)ga;
            *(uint4*)lb = *(const uint4*)gb;
#endif
        }
    };

    v8f acc[4][2] = {};

    stage(0, 0);
    vqvae_async_fence();
    __syncthreads();

    int buf = 0;
    for (int k0 = 0; k0 < K; k0 += TILE_K) {
        if (k0 + TILE_K < K) stage(k0 + TILE_K, buf ^ 1);

        #pragma unroll
        for (int kk = 0; kk < TILE_K; kk += 32) {
            // A fragment: 16-bit A 16x32 layout
            // lanes 0-15: K {0..7,16..23}; lanes 16-31: K {8..15,24..31}
            v16h af[4];
            #pragma unroll
            for (int i = 0; i < 4; ++i) {
                const _Float16* src = &Ash[buf][(mw * 64 + i * 16 + lm) * LDSROW + kk];
                #pragma unroll
                for (int v = 0; v < 8; ++v) {
                    const int kb = ((v & 4) ? 16 : 0) + kh * 8 + (v & 3) * 2;
                    af[i][2 * v]     = src[kb];
                    af[i][2 * v + 1] = src[kb + 1];
                }
            }
            // B fragment: 16-bit B 32x16 layout
            // lanes 0-15: K {0..7,16..23}; lanes 16-31: K {8..15,24..31}
            v16h bf[2];
            #pragma unroll
            for (int j = 0; j < 2; ++j) {
                const _Float16* src = &Bsh[buf][(nw * 32 + j * 16 + lm) * LDSROW + kk];
                #pragma unroll
                for (int v = 0; v < 8; ++v) {
                    const int kb = ((v & 4) ? 16 : 0) + kh * 8 + (v & 3) * 2;
                    bf[j][2 * v]     = src[kb];
                    bf[j][2 * v + 1] = src[kb + 1];
                }
            }
            #pragma unroll
            for (int i = 0; i < 4; ++i)
                #pragma unroll
                for (int j = 0; j < 2; ++j)
                    acc[i][j] = __builtin_amdgcn_wmma_f32_16x16x32_f16(
                        false, af[i], false, bf[j], (short)0, acc[i][j], false, false);
        }

        vqvae_async_fence();
        __syncthreads();
        buf ^= 1;
    }

    // epilogue: C/D layout m = v + 8*kh, n = lm
    #pragma unroll
    for (int i = 0; i < 4; ++i) {
        const int mbase = m0 + mw * 64 + i * 16 + kh * 8;
        #pragma unroll
        for (int j = 0; j < 2; ++j) {
            const int nn = n0 + nw * 32 + j * 16 + lm;
            const float bv = bias ? bias[nn] : 0.0f;
            #pragma unroll
            for (int v = 0; v < 8; ++v) {
                float r = acc[i][j][v] + bv;
                if (relu) r = fmaxf(r, 0.0f);
                if (out32) out32[(size_t)(mbase + v) * N + nn] = r;
                if (out16) out16[(size_t)(mbase + v) * N + nn] = (_Float16)r;
            }
        }
    }
}

// elementwise f32 -> f16 (vectorized by 4)
__global__ __launch_bounds__(256)
void vqvae_cvt_f16(const float* __restrict__ src, _Float16* __restrict__ dst, long n4)
{
    const long stride = (long)gridDim.x * blockDim.x;
    for (long i = (long)blockIdx.x * blockDim.x + threadIdx.x; i < n4; i += stride) {
        const float4 v = ((const float4*)src)[i];
        _Float16* d = dst + i * 4;
        d[0] = (_Float16)v.x; d[1] = (_Float16)v.y;
        d[2] = (_Float16)v.z; d[3] = (_Float16)v.w;
    }
}

// W[K,N] f32 -> Wt[N,K] f16, 32x32 LDS tiles
__global__ __launch_bounds__(256)
void vqvae_transpose_cvt(const float* __restrict__ W, _Float16* __restrict__ Wt,
                         int K, int N)
{
    __shared__ _Float16 t[32][33];
    const int n0 = blockIdx.x * 32;
    const int k0 = blockIdx.y * 32;
    const int tx = threadIdx.x, ty = threadIdx.y;   // (32, 8)
    #pragma unroll
    for (int i = 0; i < 4; ++i) {
        const int k = ty + i * 8;
        t[tx][k] = (_Float16)W[(size_t)(k0 + k) * N + n0 + tx];
    }
    __syncthreads();
    #pragma unroll
    for (int i = 0; i < 4; ++i) {
        const int nrow = ty + i * 8;
        Wt[(size_t)(n0 + nrow) * K + k0 + tx] = t[nrow][tx];
    }
}

// codebook [1024,512] f32 -> f16 copy (native layout == Bt[N=1024,K=512]) + cc[c]=||c||^2
__global__ __launch_bounds__(256)
void vqvae_cb_prep(const float* __restrict__ cb,
                   _Float16* __restrict__ cbh,
                   float* __restrict__ cc)
{
    __shared__ float red[256];
    const int c = blockIdx.x, tid = threadIdx.x;
    float s = 0.0f;
    for (int i = tid; i < 512; i += 256) {
        const float v = cb[(size_t)c * 512 + i];
        cbh[(size_t)c * 512 + i] = (_Float16)v;
        s += v * v;
    }
    red[tid] = s; __syncthreads();
    for (int off = 128; off > 0; off >>= 1) {
        if (tid < off) red[tid] += red[tid + off];
        __syncthreads();
    }
    if (tid == 0) cc[c] = red[0];
}

__global__ void vqvae_zero(float* p) { *p = 0.0f; }

// Per row: argmin_c(cc[c] - 2*S[b,c]) -> idx, z_q gather, straight-through, loss partial.
__global__ __launch_bounds__(256)
void vqvae_vq_select(const float* __restrict__ S,    // [8192,1024]
                     const float* __restrict__ cc,   // [1024]
                     const float* __restrict__ cb,   // [1024,512] f32
                     const float* __restrict__ z,    // [8192,512] f32
                     float* __restrict__ zq_st,      // [8192,512] f32
                     _Float16* __restrict__ zqh,     // [8192,512] f16 (decoder input)
                     float* __restrict__ idx_out,    // [8192] as float
                     float* __restrict__ loss_acc)
{
    __shared__ float bestd[256];
    __shared__ int   besti[256];
    const int b = blockIdx.x, tid = threadIdx.x;

    float bd = 3.402823466e+38f; int bi = 0;
    for (int c = tid; c < 1024; c += 256) {
        const float d = cc[c] - 2.0f * S[(size_t)b * 1024 + c];
        if (d < bd) { bd = d; bi = c; }   // ascending c -> first-min per thread
    }
    bestd[tid] = bd; besti[tid] = bi; __syncthreads();
    for (int off = 128; off > 0; off >>= 1) {
        if (tid < off) {
            const float d2 = bestd[tid + off]; const int i2 = besti[tid + off];
            if (d2 < bestd[tid] || (d2 == bestd[tid] && i2 < besti[tid])) {
                bestd[tid] = d2; besti[tid] = i2;
            }
        }
        __syncthreads();
    }
    const int k = besti[0];
    if (tid == 0) idx_out[b] = (float)k;

    float ls = 0.0f;
    for (int i = tid; i < 512; i += 256) {
        const float zv = z[(size_t)b * 512 + i];
        const float qv = cb[(size_t)k * 512 + i];
        const float st = zv + (qv - zv);            // straight-through forward
        zq_st[(size_t)b * 512 + i] = st;
        zqh[(size_t)b * 512 + i]   = (_Float16)st;
        const float dl = qv - zv;
        ls += dl * dl;
    }
    __syncthreads();
    bestd[tid] = ls; __syncthreads();
    for (int off = 128; off > 0; off >>= 1) {
        if (tid < off) bestd[tid] += bestd[tid + off];
        __syncthreads();
    }
    if (tid == 0) atomicAdd(loss_acc, bestd[0]);
}

__global__ void vqvae_finalize_loss(const float* acc, float* loss) {
    *loss = 1.25f * (*acc) / (8192.0f * 512.0f);   // embedding + 0.25*commitment
}

extern "C" void kernel_launch(void* const* d_in, const int* in_sizes, int n_in,
                              void* d_out, int out_size, void* d_ws, size_t ws_size,
                              hipStream_t stream) {
    (void)in_sizes; (void)n_in; (void)out_size; (void)ws_size;

    const float* x  = (const float*)d_in[0];
    const float* cb = (const float*)d_in[1];
    const float* we[4] = {(const float*)d_in[2], (const float*)d_in[4],
                          (const float*)d_in[6], (const float*)d_in[8]};
    const float* be[4] = {(const float*)d_in[3], (const float*)d_in[5],
                          (const float*)d_in[7], (const float*)d_in[9]};
    const float* wd[4] = {(const float*)d_in[10], (const float*)d_in[12],
                          (const float*)d_in[14], (const float*)d_in[16]};
    const float* bd[4] = {(const float*)d_in[11], (const float*)d_in[13],
                          (const float*)d_in[15], (const float*)d_in[17]};

    float* out = (float*)d_out;
    float* z    = out;                                   // 8192*512
    float* zq   = out + (size_t)8192 * 512;              // 8192*512
    float* pred = out + (size_t)8192 * 512 * 2;          // 8192*4096
    float* loss = out + (size_t)8192 * 512 * 2 + (size_t)8192 * 4096;  // 1
    float* idxf = loss + 1;                              // 8192

    // ---- workspace carve-up ----
    char* p = (char*)d_ws;
    auto carve = [&](size_t bytes) {
        char* r = p; p += (bytes + 255) & ~(size_t)255; return r;
    };
    _Float16* act0 = (_Float16*)carve((size_t)8192 * 4096 * 2);  // 64 MB ping
    _Float16* act1 = (_Float16*)carve((size_t)8192 * 2048 * 2);  // 32 MB pong
    float*    Smat = (float*)carve((size_t)8192 * 1024 * 4);     // 32 MB scores
    const int encK[4] = {4096, 2048, 1024, 512}, encN[4] = {2048, 1024, 512, 512};
    const int decK[4] = {512, 1024, 2048, 4096}, decN[4] = {1024, 2048, 4096, 4096};
    _Float16* wte[4]; _Float16* wtd[4];
    for (int i = 0; i < 4; ++i) wte[i] = (_Float16*)carve((size_t)encK[i] * encN[i] * 2);
    for (int i = 0; i < 4; ++i) wtd[i] = (_Float16*)carve((size_t)decK[i] * decN[i] * 2);
    _Float16* cbh = (_Float16*)carve((size_t)1024 * 512 * 2);
    float* cc   = (float*)carve(1024 * 4);
    float* lacc = (float*)carve(256);

    // ---- preprocessing: f16 conversions / weight transposes ----
    vqvae_cvt_f16<<<4096, 256, 0, stream>>>(x, act0, (long)8192 * 4096 / 4);
    for (int i = 0; i < 4; ++i) {
        dim3 g(encN[i] / 32, encK[i] / 32), b(32, 8);
        vqvae_transpose_cvt<<<g, b, 0, stream>>>(we[i], wte[i], encK[i], encN[i]);
    }
    for (int i = 0; i < 4; ++i) {
        dim3 g(decN[i] / 32, decK[i] / 32), b(32, 8);
        vqvae_transpose_cvt<<<g, b, 0, stream>>>(wd[i], wtd[i], decK[i], decN[i]);
    }
    vqvae_cb_prep<<<1024, 256, 0, stream>>>(cb, cbh, cc);
    vqvae_zero<<<1, 1, 0, stream>>>(lacc);

    auto gemm = [&](const _Float16* Am, const _Float16* Bm, const float* bm,
                    float* o32, _Float16* o16, int M, int N, int K, int relu) {
        dim3 grid(N / TILE_N, M / TILE_M);
        vqvae_gemm_wmma<<<grid, 256, 0, stream>>>(Am, Bm, bm, o32, o16, M, N, K, relu);
    };

    // encoder (f16 activations ping-pong)
    gemm(act0, wte[0], be[0], nullptr, act1, 8192, 2048, 4096, 1);  // h1
    gemm(act1, wte[1], be[1], nullptr, act0, 8192, 1024, 2048, 1);  // h2
    gemm(act0, wte[2], be[2], nullptr, act1, 8192,  512, 1024, 1);  // h3
    gemm(act1, wte[3], be[3], z,       act0, 8192,  512,  512, 0);  // z (f32+f16)

    // VQ: S = Z @ C^T (codebook native layout is Bt), then argmin/gather/loss
    gemm(act0, cbh, nullptr, Smat, nullptr, 8192, 1024, 512, 0);
    vqvae_vq_select<<<8192, 256, 0, stream>>>(Smat, cc, cb, z, zq, act1, idxf, lacc);

    // decoder
    gemm(act1, wtd[0], bd[0], nullptr, act0, 8192, 1024,  512, 1);  // g1
    gemm(act0, wtd[1], bd[1], nullptr, act1, 8192, 2048, 1024, 1);  // g2
    gemm(act1, wtd[2], bd[2], nullptr, act0, 8192, 4096, 2048, 1);  // g3
    gemm(act0, wtd[3], bd[3], pred,  nullptr, 8192, 4096, 4096, 0); // pred_x

    vqvae_finalize_loss<<<1, 1, 0, stream>>>(lacc, loss);
}
